// GraphGRUCell_60309930770883
// MI455X (gfx1250) — matrix operand order repaired and checked by reading, not compiled, not run on hardware
//
#include <hip/hip_runtime.h>
#include <hip/hip_bf16.h>

// ---------------------------------------------------------------------------
// GraphGRU fused cell for MI455X (gfx1250, wave32, WMMA).
//   B=32768, D=U=256, K=4, 3U=768.
// 5 GEMMs of [B,256]x[256,768] fused with gather/mask/activations.
// f16 WMMA (v_wmma_f32_16x16x32_f16), f32 accumulate + f32 elementwise.
// ---------------------------------------------------------------------------

typedef _Float16 v8h  __attribute__((ext_vector_type(8)));
typedef _Float16 v16h __attribute__((ext_vector_type(16)));
typedef float    v8f  __attribute__((ext_vector_type(8)));

#define B_    32768
#define D_    256
#define U_    256
#define K_    4
#define U3_   768
#define BM    64          // rows of B per workgroup
#define PADH  264         // padded LDS row length in halves (264*2=528B, 16B aligned)

// ---- pre-pass: convert weights to f16, transposed to [col][k] (k contiguous)
__global__ void prep_weights(const float* __restrict__ w,    // kernel [256][768]
                             const float* __restrict__ rk,   // recurrent [4][256][768]
                             _Float16* __restrict__ wt,      // [768][256]
                             _Float16* __restrict__ rt)      // [4][768][256]
{
    const int bi = blockIdx.x;      // 0 .. 5*768-1
    const int kk = threadIdx.x;     // 0 .. 255 (K index)
    if (bi < U3_) {
        wt[(size_t)bi * 256 + kk] = (_Float16)w[(size_t)kk * U3_ + bi];
    } else {
        const int ci  = bi - U3_;
        const int mat = ci / U3_;
        const int col = ci % U3_;
        rt[((size_t)mat * U3_ + col) * 256 + kk] =
            (_Float16)rk[((size_t)mat * 256 + kk) * U3_ + col];
    }
}

__device__ __forceinline__ float sigmoidf_(float x) {
    return 1.0f / (1.0f + __expf(-x));
}

__global__ __launch_bounds__(256)
void graph_gru_fused(const float* __restrict__ inputs,        // [B][256]
                     const float* __restrict__ states,        // [4][B][256]
                     const int*   __restrict__ etypes,        // [B][4]
                     const unsigned char* __restrict__ cmask, // [B][4] bool
                     const float* __restrict__ bias,          // [5][768]
                     const float* __restrict__ emb,           // [40][256]
                     const _Float16* __restrict__ wt,         // [768][256]
                     const _Float16* __restrict__ rt,         // [4][768][256]
                     float* __restrict__ out)                 // [B][256]
{
    __shared__ _Float16 Xs[BM][PADH];            // 33,792 B
    __shared__ _Float16 Ss[K_][BM][PADH];        // 135,168 B
    __shared__ unsigned char Msk[BM * K_];       // 256 B

    const int tid = threadIdx.x;
    const int b0  = blockIdx.x * BM;

    // ---- stage mask (BM*K_ == 256 == blockDim.x)
    Msk[tid] = cmask[(size_t)b0 * K_ + tid];

    // ---- stage X tile as f16 (64 rows x 256 cols; 16 float4 per thread)
    {
        const float4* in4 = (const float4*)(inputs + (size_t)b0 * D_);
        #pragma unroll
        for (int i = 0; i < 16; ++i) {
            const int li  = tid + i * 256;
            const int row = li >> 6;       // 64 float4 per row
            const int c4  = li & 63;
            const float4 v = in4[row * 64 + c4];
            Xs[row][c4 * 4 + 0] = (_Float16)v.x;
            Xs[row][c4 * 4 + 1] = (_Float16)v.y;
            Xs[row][c4 * 4 + 2] = (_Float16)v.z;
            Xs[row][c4 * 4 + 3] = (_Float16)v.w;
        }
    }
    __syncthreads();   // Msk visible before state staging reads it

    // ---- stage state_k = states_k * where(mask, emb[et], 1)  (f16, all 4 k)
    for (int k = 0; k < K_; ++k) {
        const float4* st4 = (const float4*)(states + ((size_t)k * B_ + b0) * U_);
        #pragma unroll
        for (int i = 0; i < 16; ++i) {
            const int li  = tid + i * 256;
            const int row = li >> 6;
            const int c4  = li & 63;
            const float4 s = st4[row * 64 + c4];
            float4 e = make_float4(1.f, 1.f, 1.f, 1.f);
            if (Msk[row * K_ + k]) {
                const int et = etypes[(size_t)(b0 + row) * K_ + k];
                e = ((const float4*)(emb + (size_t)et * U_))[c4];
            }
            Ss[k][row][c4 * 4 + 0] = (_Float16)(s.x * e.x);
            Ss[k][row][c4 * 4 + 1] = (_Float16)(s.y * e.y);
            Ss[k][row][c4 * 4 + 2] = (_Float16)(s.z * e.z);
            Ss[k][row][c4 * 4 + 3] = (_Float16)(s.w * e.w);
        }
    }
    __syncthreads();

    // ---- wave/fragment coordinates (8 waves = 4 row-frags x 2 col-frags)
    const int wave = tid >> 5;
    const int lane = tid & 31;
    const int rw   = wave & 3;        // row fragment (16 rows each)
    const int cw   = wave >> 2;       // column fragment within 32-col u-tile
    const int rb   = rw * 16;
    const int m    = lane & 15;       // A row / B,C column within fragment
    const int hi   = lane >> 4;       // lane half

    // per-lane row mask bits (C-frag rows: rb + hi*8 + i)
    unsigned mk[K_];
    #pragma unroll
    for (int k = 0; k < K_; ++k) {
        unsigned v = 0;
        #pragma unroll
        for (int i = 0; i < 8; ++i)
            v |= (unsigned)(Msk[(rb + hi * 8 + i) * K_ + k] != 0) << i;
        mk[k] = v;
    }

    for (int ut = 0; ut < U_ / 32; ++ut) {
        const int cb   = ut * 32 + cw * 16;
        const int colx = cb + m;                 // this lane's output column u

        // -------- input projection: xz/xr/xh fragments (K=256 in 8 steps)
        v8f xz = {0,0,0,0,0,0,0,0};
        v8f xr = {0,0,0,0,0,0,0,0};
        v8f xh = {0,0,0,0,0,0,0,0};
        #pragma unroll
        for (int kk = 0; kk < 8; ++kk) {
            const int k0 = kk * 32;
            const v8h alo = *(const v8h*)&Xs[rb + m][k0 + hi * 16];
            const v8h ahi = *(const v8h*)&Xs[rb + m][k0 + hi * 16 + 8];
            const v16h a = __builtin_shufflevector(alo, ahi,
                0,1,2,3,4,5,6,7,8,9,10,11,12,13,14,15);
            const _Float16* wb = wt + (size_t)colx * 256 + k0 + 16 * hi;
            const v16h bz = *(const v16h*)(wb);
            const v16h br = *(const v16h*)(wb + 256u * 256u);  // +256 columns
            const v16h bh = *(const v16h*)(wb + 512u * 256u);  // +512 columns
            xz = __builtin_amdgcn_wmma_f32_16x16x32_f16(false, a, false, bz, (short)0, xz, false, false);
            xr = __builtin_amdgcn_wmma_f32_16x16x32_f16(false, a, false, br, (short)0, xr, false, false);
            xh = __builtin_amdgcn_wmma_f32_16x16x32_f16(false, a, false, bh, (short)0, xh, false, false);
        }
        // fold bias[0] in (per-lane scalar: column is fixed per lane)
        {
            const float b0z = bias[colx];
            const float b0r = bias[256 + colx];
            const float b0h = bias[512 + colx];
            #pragma unroll
            for (int i = 0; i < 8; ++i) { xz[i] += b0z; xr[i] += b0r; xh[i] += b0h; }
        }

        // -------- k loop: recurrent GEMM frags + masked combine
        v8f ah  = {0,0,0,0,0,0,0,0};
        v8f az  = {0,0,0,0,0,0,0,0};
        v8f azh = {0,0,0,0,0,0,0,0};
        for (int k = 0; k < K_; ++k) {
            v8f rz = {0,0,0,0,0,0,0,0};
            v8f rr = {0,0,0,0,0,0,0,0};
            v8f rh = {0,0,0,0,0,0,0,0};
            const _Float16* rbase = rt + ((size_t)k * U3_ + colx) * 256;
            #pragma unroll
            for (int kk = 0; kk < 8; ++kk) {
                const int k0 = kk * 32;
                const v8h alo = *(const v8h*)&Ss[k][rb + m][k0 + hi * 16];
                const v8h ahi = *(const v8h*)&Ss[k][rb + m][k0 + hi * 16 + 8];
                const v16h a = __builtin_shufflevector(alo, ahi,
                    0,1,2,3,4,5,6,7,8,9,10,11,12,13,14,15);
                const _Float16* p = rbase + k0 + 16 * hi;
                const v16h bz = *(const v16h*)(p);
                const v16h br = *(const v16h*)(p + 65536);
                const v16h bh = *(const v16h*)(p + 131072);
                rz = __builtin_amdgcn_wmma_f32_16x16x32_f16(false, a, false, bz, (short)0, rz, false, false);
                rr = __builtin_amdgcn_wmma_f32_16x16x32_f16(false, a, false, br, (short)0, rr, false, false);
                rh = __builtin_amdgcn_wmma_f32_16x16x32_f16(false, a, false, bh, (short)0, rh, false, false);
            }
            const float bkz = bias[(1 + k) * U3_ + colx];
            const float bkr = bias[(1 + k) * U3_ + 256 + colx];
            const float bkh = bias[(1 + k) * U3_ + 512 + colx];
            const unsigned mkk = mk[k];
            #pragma unroll
            for (int i = 0; i < 8; ++i) {
                const float mf = (float)((mkk >> i) & 1u);
                const float z  = sigmoidf_(xz[i] + rz[i] + bkz);
                const float r  = sigmoidf_(xr[i] + rr[i] + bkr);
                const float s  = (float)Ss[k][rb + hi * 8 + i][colx];
                ah[i]  += mf * (r * (rh[i] + bkh));
                az[i]  += mf * z;
                azh[i] += mf * z * s;
            }
        }

        // -------- final combine + store h (inv_k = 1/4)
        #pragma unroll
        for (int i = 0; i < 8; ++i) {
            const float hh = tanhf(xh[i] + ah[i] * 0.25f);
            const float h  = (1.0f - az[i] * 0.25f) * hh + azh[i] * 0.25f;
            out[(size_t)(b0 + rb + hi * 8 + i) * U_ + colx] = h;
        }
    }
}

extern "C" void kernel_launch(void* const* d_in, const int* in_sizes, int n_in,
                              void* d_out, int out_size, void* d_ws, size_t ws_size,
                              hipStream_t stream) {
    (void)in_sizes; (void)n_in; (void)out_size; (void)ws_size;
    const float*         inputs = (const float*)d_in[0];
    const float*         states = (const float*)d_in[1];
    const int*           etypes = (const int*)d_in[2];
    const unsigned char* cmask  = (const unsigned char*)d_in[3];  // jnp bool = 1 byte
    const float*         kern   = (const float*)d_in[4];
    const float*         rkern  = (const float*)d_in[5];
    const float*         bias   = (const float*)d_in[6];
    const float*         emb    = (const float*)d_in[7];
    float*               out    = (float*)d_out;

    _Float16* wt = (_Float16*)d_ws;                   // [768][256]  f16
    _Float16* rt = wt + (size_t)U3_ * 256;            // [4][768][256] f16 (~2 MB total)

    prep_weights<<<U3_ * 5, 256, 0, stream>>>(kern, rkern, wt, rt);
    graph_gru_fused<<<B_ / BM, 256, 0, stream>>>(inputs, states, etypes, cmask,
                                                 bias, emb, wt, rt, out);
}
